// SeqLSTMv2_73564199846361
// MI455X (gfx1250) — matrix operand order repaired and checked by reading, not compiled, not run on hardware
//
#include <hip/hip_runtime.h>
#include <math.h>

#define VOCAB 32000
#define EDIN  512
#define HD    1024
#define BB    32
#define SS    64
#define TT    64

typedef __attribute__((ext_vector_type(16))) __bf16 v16bf;
typedef __attribute__((ext_vector_type(8)))  float  v8f;

union FragAB { v16bf v; unsigned u[8]; };

__device__ __forceinline__ unsigned short f2bf(float f) {
  unsigned u = __float_as_uint(f);
  u += 0x7FFFu + ((u >> 16) & 1u);   // round-to-nearest-even
  return (unsigned short)(u >> 16);
}

__device__ __forceinline__ float sigmoidf_(float x) { return 1.0f / (1.0f + expf(-x)); }

// ---------------- utility / elementwise kernels ----------------

__global__ void cvt_bf16_kernel(const float* __restrict__ src, unsigned short* __restrict__ dst, long n) {
  long i = (long)blockIdx.x * blockDim.x + threadIdx.x;
  if (i < n) dst[i] = f2bf(src[i]);
}

__global__ void add_vec_kernel(const float* __restrict__ a, const float* __restrict__ b,
                               float* __restrict__ o, int n) {
  int i = blockIdx.x * blockDim.x + threadIdx.x;
  if (i < n) o[i] = a[i] + b[i];
}

__global__ void zero_f32_kernel(float* p, long n) {
  long i = (long)blockIdx.x * blockDim.x + threadIdx.x;
  if (i < n) p[i] = 0.0f;
}

__global__ void zero_u16_kernel(unsigned short* p, long n) {
  long i = (long)blockIdx.x * blockDim.x + threadIdx.x;
  if (i < n) p[i] = 0;
}

__global__ void zero_out_t0_kernel(float* __restrict__ out) {
  int i = blockIdx.x * blockDim.x + threadIdx.x;   // BB*VOCAB elements
  if (i < BB * VOCAB) {
    int b = i / VOCAB, v = i % VOCAB;
    out[(size_t)b * TT * VOCAB + v] = 0.0f;
  }
}

// x_bf laid out time-major: [t][b][d], bf16
__global__ void embed_src_kernel(const int* __restrict__ src, const float* __restrict__ emb,
                                 unsigned short* __restrict__ xbf) {
  int i = blockIdx.x * blockDim.x + threadIdx.x;
  if (i >= SS * BB * EDIN) return;
  int d  = i % EDIN;
  int tb = i / EDIN;           // t*BB + b
  int b  = tb % BB;
  int t  = tb / BB;
  int tokid = src[b * SS + t];
  xbf[i] = f2bf(emb[(size_t)tokid * EDIN + d]);
}

__global__ void embed_tok_kernel(const int* __restrict__ tok, const float* __restrict__ emb,
                                 unsigned short* __restrict__ ebf) {
  int i = blockIdx.x * blockDim.x + threadIdx.x;
  if (i >= BB * HD) return;
  int b = i / HD, j = i % HD;
  ebf[i] = f2bf(emb[(size_t)tok[b] * HD + j]);
}

// Reproduce reference's stack([hf,hb]).reshape(1,B,2H)[-1] gather, emit bf16
__global__ void gather_hc_kernel(const float* __restrict__ hf, const float* __restrict__ hb,
                                 unsigned short* __restrict__ hv) {
  int i = blockIdx.x * blockDim.x + threadIdx.x;   // BB * 2*HD
  if (i >= BB * 2 * HD) return;
  int b = i / (2 * HD);
  int j = i % (2 * HD);
  int half = j / HD, jj = j % HD;
  const float* s = (b < BB / 2) ? hf : hb;
  int sb = 2 * (b % (BB / 2)) + half;
  hv[i] = f2bf(s[(size_t)sb * HD + jj]);
}

__global__ void lstm_cell_kernel(const float* __restrict__ gates, float* __restrict__ c,
                                 float* __restrict__ h, unsigned short* __restrict__ hbf) {
  int i = blockIdx.x * blockDim.x + threadIdx.x;   // BB*HD
  if (i >= BB * HD) return;
  int b = i / HD, j = i % HD;
  const float* g = gates + (size_t)b * 4 * HD;
  float ig = sigmoidf_(g[j]);
  float fg = sigmoidf_(g[HD + j]);
  float gg = tanhf(g[2 * HD + j]);
  float og = sigmoidf_(g[3 * HD + j]);
  float cn = fg * c[i] + ig * gg;
  c[i] = cn;
  float hn = og * tanhf(cn);
  h[i] = hn;
  hbf[i] = f2bf(hn);
}

__global__ void init_tok_kernel(const int* __restrict__ trg, int* __restrict__ tok) {
  int b = threadIdx.x;
  if (b < BB) tok[b] = trg[b * TT];
}

__global__ void argmax_next_kernel(const float* __restrict__ logits, long ldc,
                                   const int* __restrict__ trg, const int* __restrict__ tfm,
                                   int t, int* __restrict__ tok) {
  __shared__ float smax[256];
  __shared__ int   sidx[256];
  int b = blockIdx.x;
  const float* row = logits + (size_t)b * ldc;
  float best = -3.4e38f;
  int bi = 0;
  for (int v = threadIdx.x; v < VOCAB; v += 256) {
    float x = row[v];
    if (x > best) { best = x; bi = v; }   // strict > keeps first occurrence
  }
  smax[threadIdx.x] = best; sidx[threadIdx.x] = bi;
  __syncthreads();
  for (int s = 128; s > 0; s >>= 1) {
    if ((int)threadIdx.x < s) {
      float o = smax[threadIdx.x + s]; int oi = sidx[threadIdx.x + s];
      if (o > smax[threadIdx.x] || (o == smax[threadIdx.x] && oi < sidx[threadIdx.x])) {
        smax[threadIdx.x] = o; sidx[threadIdx.x] = oi;
      }
    }
    __syncthreads();
  }
  if (threadIdx.x == 0) {
    tok[b] = (tfm[t] != 0) ? trg[b * TT + (t + 1)] : sidx[0];
  }
}

// ---------------- WMMA bf16 GEMM: C = A*W^T (+ A2*W2^T) (+bias[n]) (+addend[m,n]) ---------
// A: M x K bf16 row-major (lda == K). W: N x K bf16 row-major (ldw == K) -> computes A @ W.T
// One wave per 16x16 output tile; K in steps of 32 via V_WMMA_F32_16X16X32_BF16.
__global__ void wmma_gemm_kernel(
    const unsigned short* __restrict__ A,
    const unsigned short* __restrict__ W,
    const unsigned short* __restrict__ A2,
    const unsigned short* __restrict__ W2,
    const float* __restrict__ bias,
    const float* __restrict__ addend,
    float* __restrict__ C, long ldc,
    unsigned short* __restrict__ Cbf,
    int M, int N, int K)
{
  int lane = threadIdx.x & 31;
  int wave = threadIdx.x >> 5;
  int ntN = N >> 4;
  int ntiles = (M >> 4) * ntN;
  int tile = blockIdx.x * (blockDim.x >> 5) + wave;
  if (tile >= ntiles) return;              // whole-wave uniform; EXEC stays all-ones for WMMA
  int m0 = (tile / ntN) << 4;
  int n0 = (tile % ntN) << 4;
  int g = lane >> 4;                       // half-wave group
  int r = lane & 15;                       // A row within tile / B-C column within tile

  v8f acc = {0.f, 0.f, 0.f, 0.f, 0.f, 0.f, 0.f, 0.f};

  for (int pass = 0; pass < 2; ++pass) {
    const unsigned short* Ap = pass ? A2 : A;
    const unsigned short* Wp = pass ? W2 : W;
    if (!Ap) break;
    // dword views: each dword holds bf16 pair (k, k+1), k even -- contiguous in row-major
    const unsigned* Arow = (const unsigned*)(Ap + (size_t)(m0 + r) * K);
    const unsigned* Wrow = (const unsigned*)(Wp + (size_t)(n0 + r) * K);
    for (int k0 = 0; k0 < K; k0 += 32) {
      int kb = k0 >> 1;
      // A 16x32 bf16 layout: lanes 0-15 M=0..15; vgpr0-3: K=2v+8g, vgpr4-7: K=16+2(v-4)+8g
      const uint4 a0 = *(const uint4*)(Arow + kb + 4 * g);
      const uint4 a1 = *(const uint4*)(Arow + kb + 8 + 4 * g);
      // B 32x16 bf16 layout: lane n; vgpr0-3: K=2v+8g, vgpr4-7: K=16+2(v-4)+8g
      const uint4 b0 = *(const uint4*)(Wrow + kb + 4 * g);
      const uint4 b1 = *(const uint4*)(Wrow + kb + 8 + 4 * g);
      FragAB a, b;
      a.u[0] = a0.x; a.u[1] = a0.y; a.u[2] = a0.z; a.u[3] = a0.w;
      a.u[4] = a1.x; a.u[5] = a1.y; a.u[6] = a1.z; a.u[7] = a1.w;
      b.u[0] = b0.x; b.u[1] = b0.y; b.u[2] = b0.z; b.u[3] = b0.w;
      b.u[4] = b1.x; b.u[5] = b1.y; b.u[6] = b1.z; b.u[7] = b1.w;
      if (k0 + 32 < K) __builtin_prefetch(Wrow + kb + 16, 0, 1);  // global_prefetch_b8
      acc = __builtin_amdgcn_wmma_f32_16x16x32_bf16(
          false, a.v, false, b.v, (short)0, acc, false, false);
    }
  }

  // C/D layout: vgpr rr -> M = rr + 8g, N = r
  int n = n0 + r;
  float bn = bias ? bias[n] : 0.0f;
  for (int rr = 0; rr < 8; ++rr) {
    int m = m0 + rr + 8 * g;
    float val = acc[rr] + bn;
    if (addend) val += addend[(size_t)m * N + n];
    C[(size_t)m * ldc + n] = val;
    if (Cbf) Cbf[(size_t)m * ldc + n] = f2bf(val);
  }
}

// ---------------- host orchestration ----------------

extern "C" void kernel_launch(void* const* d_in, const int* in_sizes, int n_in,
                              void* d_out, int out_size, void* d_ws, size_t ws_size,
                              hipStream_t stream) {
  (void)in_sizes; (void)n_in; (void)out_size; (void)ws_size;

  const int*   src     = (const int*)d_in[0];
  const int*   trg     = (const int*)d_in[1];
  const int*   tfm     = (const int*)d_in[2];
  const float* in_emb  = (const float*)d_in[3];
  const float* out_emb = (const float*)d_in[4];
  const float* Wih_f   = (const float*)d_in[5];
  const float* Whh_f   = (const float*)d_in[6];
  const float* bih_f   = (const float*)d_in[7];
  const float* bhh_f   = (const float*)d_in[8];
  const float* Wih_b   = (const float*)d_in[9];
  const float* Whh_b   = (const float*)d_in[10];
  const float* bih_b   = (const float*)d_in[11];
  const float* bhh_b   = (const float*)d_in[12];
  const float* Wht     = (const float*)d_in[13];
  const float* Wct     = (const float*)d_in[14];
  const float* dWih    = (const float*)d_in[15];
  const float* dWhh    = (const float*)d_in[16];
  const float* dbih    = (const float*)d_in[17];
  const float* dbhh    = (const float*)d_in[18];
  const float* fcW     = (const float*)d_in[19];
  const float* oW      = (const float*)d_in[20];
  float* out = (float*)d_out;

  char* base = (char*)d_ws;
  size_t off = 0;
  auto alloc = [&](size_t bytes) -> void* {
    void* p = base + off;
    off += (bytes + 255) & ~(size_t)255;
    return p;
  };

  unsigned short* Wih_f_bf = (unsigned short*)alloc((size_t)4 * HD * EDIN * 2);
  unsigned short* Whh_f_bf = (unsigned short*)alloc((size_t)4 * HD * HD * 2);
  unsigned short* Wih_b_bf = (unsigned short*)alloc((size_t)4 * HD * EDIN * 2);
  unsigned short* Whh_b_bf = (unsigned short*)alloc((size_t)4 * HD * HD * 2);
  unsigned short* dWih_bf  = (unsigned short*)alloc((size_t)4 * HD * HD * 2);
  unsigned short* dWhh_bf  = (unsigned short*)alloc((size_t)4 * HD * HD * 2);
  unsigned short* Wht_bf   = (unsigned short*)alloc((size_t)HD * 2 * HD * 2);
  unsigned short* Wct_bf   = (unsigned short*)alloc((size_t)HD * 2 * HD * 2);
  unsigned short* fcW_bf   = (unsigned short*)alloc((size_t)HD * HD * 2);
  unsigned short* oW_bf    = (unsigned short*)alloc((size_t)VOCAB * HD * 2);
  unsigned short* x_bf     = (unsigned short*)alloc((size_t)SS * BB * EDIN * 2);
  float* xpre_f = (float*)alloc((size_t)SS * BB * 4 * HD * 4);
  float* xpre_b = (float*)alloc((size_t)SS * BB * 4 * HD * 4);
  float* bias_f = (float*)alloc((size_t)4 * HD * 4);
  float* bias_b = (float*)alloc((size_t)4 * HD * 4);
  float* bias_d = (float*)alloc((size_t)4 * HD * 4);
  float* hf = (float*)alloc((size_t)BB * HD * 4);
  float* cf = (float*)alloc((size_t)BB * HD * 4);
  float* hb = (float*)alloc((size_t)BB * HD * 4);
  float* cb = (float*)alloc((size_t)BB * HD * 4);
  unsigned short* hf_bf = (unsigned short*)alloc((size_t)BB * HD * 2);
  unsigned short* hb_bf = (unsigned short*)alloc((size_t)BB * HD * 2);
  unsigned short* hv_bf = (unsigned short*)alloc((size_t)BB * 2 * HD * 2);
  unsigned short* cv_bf = (unsigned short*)alloc((size_t)BB * 2 * HD * 2);
  float* hdec = (float*)alloc((size_t)BB * HD * 4);
  float* cdec = (float*)alloc((size_t)BB * HD * 4);
  unsigned short* hdec_bf = (unsigned short*)alloc((size_t)BB * HD * 2);
  unsigned short* e_bf    = (unsigned short*)alloc((size_t)BB * HD * 2);
  float* gates  = (float*)alloc((size_t)BB * 4 * HD * 4);
  float* gates2 = (float*)alloc((size_t)BB * 4 * HD * 4);
  float* zbuf   = (float*)alloc((size_t)BB * HD * 4);
  unsigned short* z_bf = (unsigned short*)alloc((size_t)BB * HD * 2);
  int* tok = (int*)alloc((size_t)BB * 4);

  auto cvt = [&](const float* s, unsigned short* d, long n) {
    cvt_bf16_kernel<<<(int)((n + 255) / 256), 256, 0, stream>>>(s, d, n);
  };
  auto gemm = [&](const unsigned short* A, const unsigned short* W,
                  const unsigned short* A2, const unsigned short* W2,
                  const float* bias, const float* addend,
                  float* C, long ldc, unsigned short* Cbf,
                  int M, int N, int K) {
    int tiles = (M / 16) * (N / 16);
    int blocks = (tiles + 7) / 8;           // 8 waves (16x16 tiles) per 256-thread block
    wmma_gemm_kernel<<<blocks, 256, 0, stream>>>(A, W, A2, W2, bias, addend, C, ldc, Cbf, M, N, K);
  };

  // --- weight conversion to bf16 (L2-resident working set; oW_bf = 64MB) ---
  cvt(Wih_f, Wih_f_bf, 4L * HD * EDIN);
  cvt(Whh_f, Whh_f_bf, 4L * HD * HD);
  cvt(Wih_b, Wih_b_bf, 4L * HD * EDIN);
  cvt(Whh_b, Whh_b_bf, 4L * HD * HD);
  cvt(dWih,  dWih_bf,  4L * HD * HD);
  cvt(dWhh,  dWhh_bf,  4L * HD * HD);
  cvt(Wht,   Wht_bf,   (long)HD * 2 * HD);
  cvt(Wct,   Wct_bf,   (long)HD * 2 * HD);
  cvt(fcW,   fcW_bf,   (long)HD * HD);
  cvt(oW,    oW_bf,    (long)VOCAB * HD);

  int bthr = (4 * HD + 255) / 256;
  add_vec_kernel<<<bthr, 256, 0, stream>>>(bih_f, bhh_f, bias_f, 4 * HD);
  add_vec_kernel<<<bthr, 256, 0, stream>>>(bih_b, bhh_b, bias_b, 4 * HD);
  add_vec_kernel<<<bthr, 256, 0, stream>>>(dbih,  dbhh,  bias_d, 4 * HD);

  long nhc = (long)BB * HD;
  int  ghc = (int)((nhc + 255) / 256);
  zero_f32_kernel<<<ghc, 256, 0, stream>>>(hf, nhc);
  zero_f32_kernel<<<ghc, 256, 0, stream>>>(cf, nhc);
  zero_f32_kernel<<<ghc, 256, 0, stream>>>(hb, nhc);
  zero_f32_kernel<<<ghc, 256, 0, stream>>>(cb, nhc);
  zero_u16_kernel<<<ghc, 256, 0, stream>>>(hf_bf, nhc);
  zero_u16_kernel<<<ghc, 256, 0, stream>>>(hb_bf, nhc);
  zero_out_t0_kernel<<<(BB * VOCAB + 255) / 256, 256, 0, stream>>>(out);

  // --- batched input projection: xpre[t,b,:] = x@Wih.T + (bih+bhh), both directions ---
  embed_src_kernel<<<(SS * BB * EDIN + 255) / 256, 256, 0, stream>>>(src, in_emb, x_bf);
  gemm(x_bf, Wih_f_bf, nullptr, nullptr, bias_f, nullptr, xpre_f, 4 * HD, nullptr, SS * BB, 4 * HD, EDIN);
  gemm(x_bf, Wih_b_bf, nullptr, nullptr, bias_b, nullptr, xpre_b, 4 * HD, nullptr, SS * BB, 4 * HD, EDIN);

  // --- encoder recurrence (serial): gates = h@Whh.T + xpre_t ---
  for (int t = 0; t < SS; ++t) {
    gemm(hf_bf, Whh_f_bf, nullptr, nullptr, nullptr, xpre_f + (size_t)t * BB * 4 * HD,
         gates, 4 * HD, nullptr, BB, 4 * HD, HD);
    lstm_cell_kernel<<<ghc, 256, 0, stream>>>(gates, cf, hf, hf_bf);
    gemm(hb_bf, Whh_b_bf, nullptr, nullptr, nullptr, xpre_b + (size_t)(SS - 1 - t) * BB * 4 * HD,
         gates2, 4 * HD, nullptr, BB, 4 * HD, HD);
    lstm_cell_kernel<<<ghc, 256, 0, stream>>>(gates2, cb, hb, hb_bf);
  }

  // --- bridge: hv/cv gather (reference's reshape quirk), then hidden/cell projections ---
  int g2h = (BB * 2 * HD + 255) / 256;
  gather_hc_kernel<<<g2h, 256, 0, stream>>>(hf, hb, hv_bf);
  gather_hc_kernel<<<g2h, 256, 0, stream>>>(cf, cb, cv_bf);
  gemm(hv_bf, Wht_bf, nullptr, nullptr, nullptr, nullptr, hdec, HD, hdec_bf, BB, HD, 2 * HD);
  gemm(cv_bf, Wct_bf, nullptr, nullptr, nullptr, nullptr, cdec, HD, nullptr,  BB, HD, 2 * HD);
  init_tok_kernel<<<1, BB, 0, stream>>>(trg, tok);

  // --- decoder (serial: argmax feeds next token) ---
  for (int t = 0; t < TT - 1; ++t) {
    embed_tok_kernel<<<ghc, 256, 0, stream>>>(tok, out_emb, e_bf);
    gemm(e_bf, dWih_bf, hdec_bf, dWhh_bf, bias_d, nullptr, gates, 4 * HD, nullptr, BB, 4 * HD, HD);
    lstm_cell_kernel<<<ghc, 256, 0, stream>>>(gates, cdec, hdec, hdec_bf);
    gemm(hdec_bf, fcW_bf, nullptr, nullptr, nullptr, nullptr, zbuf, HD, z_bf, BB, HD, HD);
    gemm(z_bf, oW_bf, nullptr, nullptr, nullptr, nullptr,
         out + (size_t)(t + 1) * VOCAB, (long)TT * VOCAB, nullptr, BB, VOCAB, HD);
    argmax_next_kernel<<<BB, 256, 0, stream>>>(out + (size_t)(t + 1) * VOCAB,
                                               (long)TT * VOCAB, trg, tfm, t, tok);
  }
}
